// SelfAttention_18537078850137
// MI455X (gfx1250) — matrix operand order repaired and checked by reading, not compiled, not run on hardware
//
#include <hip/hip_runtime.h>
#include <hip/hip_bf16.h>

#define NN 8192
#define DD 1024
#define HH 1024
#define GG 2048

typedef __attribute__((ext_vector_type(16))) __bf16 v16bf;
typedef __attribute__((ext_vector_type(8)))  float  v8f;

union FragB { unsigned u[8]; v16bf v; };

__device__ __forceinline__ unsigned short f32_to_bf16(float f) {
    unsigned u = __float_as_uint(f);
    unsigned r = u + 0x7FFFu + ((u >> 16) & 1u);   // round-to-nearest-even
    return (unsigned short)(r >> 16);
}
__device__ __forceinline__ unsigned pack_bf16x2(float a, float b) {
    return (unsigned)f32_to_bf16(a) | ((unsigned)f32_to_bf16(b) << 16);
}
__device__ __forceinline__ unsigned enc_f32(float f) {  // order-preserving uint encoding
    unsigned u = __float_as_uint(f);
    return (u & 0x80000000u) ? ~u : (u | 0x80000000u);
}
__device__ __forceinline__ float dec_f32(unsigned e) {
    return __uint_as_float((e & 0x80000000u) ? (e ^ 0x80000000u) : ~e);
}
__device__ __forceinline__ float wave_sum(float a) {
    #pragma unroll
    for (int off = 16; off > 0; off >>= 1) a += __shfl_xor(a, off, 32);
    return a;
}

// ---- K0: zero accumulators -------------------------------------------------
__global__ void k_init(float* xtw, float* sumw, unsigned* mg, float* zg) {
    int i = blockIdx.x * 256 + threadIdx.x;
    if (i < DD) xtw[i] = 0.f;
    if (i < GG) { mg[i] = 0u; zg[i] = 0.f; }
    if (i == 0) sumw[0] = 0.f;
}

// ---- K1: xtw[d] = sum_n w[n]*X[n,d];  sumw = sum_n w[n] --------------------
__global__ void k_xtw(const float* __restrict__ X, const float* __restrict__ w,
                      float* xtw, float* sumw) {
    int tid = threadIdx.x, n0 = blockIdx.x * 128;
    float r0 = 0.f, r1 = 0.f, r2 = 0.f, r3 = 0.f;
    for (int i = 0; i < 128; ++i) {
        int n = n0 + i;
        float wn = w[n];
        const float* xr = X + (size_t)n * DD;
        r0 += wn * xr[tid];
        r1 += wn * xr[tid + 256];
        r2 += wn * xr[tid + 512];
        r3 += wn * xr[tid + 768];
    }
    atomicAdd(xtw + tid,       r0);
    atomicAdd(xtw + tid + 256, r1);
    atomicAdd(xtw + tid + 512, r2);
    atomicAdd(xtw + tid + 768, r3);
    if (tid == 0) {
        float t = 0.f;
        for (int i = 0; i < 128; ++i) t += w[n0 + i];
        atomicAdd(sumw, t);
    }
}

// ---- K2: kw[h] = Wk[h,:].xtw + bk[h]*sumw  (one wave per h) ----------------
__global__ void k_kw(const float* __restrict__ Wk, const float* __restrict__ bk,
                     const float* __restrict__ xtw, const float* __restrict__ sumw,
                     float* kw) {
    int h = blockIdx.x * 8 + (threadIdx.x >> 5);
    int lane = threadIdx.x & 31;
    const float* row = Wk + (size_t)h * DD;
    float acc = 0.f;
    for (int d = lane; d < DD; d += 32) acc += row[d] * xtw[d];
    acc = wave_sum(acc);
    if (lane == 0) kw[h] = acc + bk[h] * sumw[0];
}

// ---- K3: u[d] = sum_h Wq[h,d]*kw[h];  cscal = bq.kw + score_b --------------
__global__ void k_u(const float* __restrict__ Wq, const float* __restrict__ bq,
                    const float* __restrict__ kw, const float* __restrict__ sb,
                    float* u, float* cscal) {
    int d = blockIdx.x * 256 + threadIdx.x;
    float acc = 0.f;
    for (int h = 0; h < HH; ++h) acc += Wq[(size_t)h * DD + d] * kw[h];
    u[d] = acc;
    if (blockIdx.x == 0 && threadIdx.x < 32) {
        float a = 0.f;
        for (int h = threadIdx.x; h < HH; h += 32) a += bq[h] * kw[h];
        a = wave_sum(a);
        if (threadIdx.x == 0) cscal[0] = a + sb[0];
    }
}

// ---- K4: s[n] = X[n,:].u + cscal  (one wave per n) -------------------------
__global__ void k_s(const float* __restrict__ X, const float* __restrict__ u,
                    const float* __restrict__ cscal, float* s) {
    int n = blockIdx.x * 8 + (threadIdx.x >> 5);
    int lane = threadIdx.x & 31;
    const float* xr = X + (size_t)n * DD;
    float acc = 0.f;
    for (int d = lane; d < DD; d += 32) acc += xr[d] * u[d];
    acc = wave_sum(acc);
    if (lane == 0) s[n] = acc + cscal[0];
}

// ---- K5: per-group masked max (encoded-uint atomicMax) ---------------------
__global__ void k_gmax(const int* __restrict__ mask, const float* __restrict__ s,
                       unsigned* mg) {
    __shared__ float sl[256];
    int tid = threadIdx.x;
    int g = blockIdx.x * 256 + tid;
    int n0 = blockIdx.y * 256;
    sl[tid] = s[n0 + tid];
    __syncthreads();
    unsigned mloc = 0u;
    for (int i = 0; i < 256; ++i) {
        int mk = mask[(size_t)(n0 + i) * GG + g];     // coalesced across tid
        if (mk != 0) {
            unsigned e = enc_f32(sl[i]);
            mloc = (e > mloc) ? e : mloc;
        }
    }
    if (mloc) atomicMax(mg + g, mloc);
}

// ---- K6: e = exp(s-m) (tile-transposed write into aw) + atomic z -----------
__global__ void k_ewrite(const int* __restrict__ mask, const float* __restrict__ s,
                         const unsigned* __restrict__ mg, float* aw, float* zg) {
    __shared__ float tileT[64][65];
    __shared__ float sv[64], mvals[64], zp[64];
    __shared__ unsigned amf[64];
    int tid = threadIdx.x;
    int g0 = blockIdx.x * 64, n0 = blockIdx.y * 64;
    if (tid < 64) {
        sv[tid] = s[n0 + tid];
        unsigned e = mg[g0 + tid];
        amf[tid] = (e == 0u);
        mvals[tid] = (e == 0u) ? 0.f : dec_f32(e);
        zp[tid] = 0.f;
    }
    __syncthreads();
    #pragma unroll
    for (int i = 0; i < 16; ++i) {
        int e = (i << 8) + tid;
        int n = e >> 6, g = e & 63;
        int mk = mask[(size_t)(n0 + n) * GG + (g0 + g)];  // g fastest -> coalesced
        float v = 0.f;
        if (mk != 0 && !amf[g]) v = __expf(sv[n] - mvals[g]);
        tileT[g][n] = v;
        if (v > 0.f) atomicAdd(&zp[g], v);
    }
    __syncthreads();
    #pragma unroll
    for (int i = 0; i < 16; ++i) {
        int e = (i << 8) + tid;
        int g = e >> 6, n = e & 63;                       // n fastest -> coalesced
        aw[(size_t)(g0 + g) * NN + (n0 + n)] = tileT[g][n];
    }
    if (tid < 64 && zp[tid] > 0.f) atomicAdd(zg + g0 + tid, zp[tid]);
}

// ---- K7: normalize aw rows; all-masked rows -> uniform 1/N -----------------
__global__ void k_final(float* aw, const float* __restrict__ zg,
                        const unsigned* __restrict__ mg) {
    int g = blockIdx.x;
    bool allm = (mg[g] == 0u);
    float invz = allm ? 0.f : 1.0f / zg[g];
    float* row = aw + (size_t)g * NN;
    for (int n = threadIdx.x; n < NN; n += 256)
        row[n] = allm ? (1.0f / (float)NN) : row[n] * invz;
}

// ---- vectorized f32->bf16 tile stagers -------------------------------------
// A tile: 128 rows x 32 k (ushort). thread t handles 16 consecutive elems.
__device__ __forceinline__ void stage_A_f32(const float* __restrict__ src,
                                            size_t row_stride, int m0, int k0,
                                            unsigned short* As, int tid) {
    int r = tid >> 1, c = (tid & 1) * 16;
    const float4* p = reinterpret_cast<const float4*>(src + (size_t)(m0 + r) * row_stride + k0 + c);
    float4 f0 = p[0], f1 = p[1], f2 = p[2], f3 = p[3];
    uint4 o0, o1;
    o0.x = pack_bf16x2(f0.x, f0.y); o0.y = pack_bf16x2(f0.z, f0.w);
    o0.z = pack_bf16x2(f1.x, f1.y); o0.w = pack_bf16x2(f1.z, f1.w);
    o1.x = pack_bf16x2(f2.x, f2.y); o1.y = pack_bf16x2(f2.z, f2.w);
    o1.z = pack_bf16x2(f3.x, f3.y); o1.w = pack_bf16x2(f3.z, f3.w);
    uint4* dst = reinterpret_cast<uint4*>(&As[tid * 16]);
    dst[0] = o0; dst[1] = o1;
}
// B tile: 64 rows x 32 k (ushort). thread t handles 8 consecutive elems.
__device__ __forceinline__ void stage_B_f32(const float* __restrict__ src,
                                            size_t row_stride, int n0, int k0,
                                            unsigned short* Bs, int tid) {
    int hl = tid >> 2, kl = (tid & 3) * 8;
    const float4* p = reinterpret_cast<const float4*>(src + (size_t)(n0 + hl) * row_stride + k0 + kl);
    float4 f0 = p[0], f1 = p[1];
    uint4 o;
    o.x = pack_bf16x2(f0.x, f0.y); o.y = pack_bf16x2(f0.z, f0.w);
    o.z = pack_bf16x2(f1.x, f1.y); o.w = pack_bf16x2(f1.z, f1.w);
    *reinterpret_cast<uint4*>(&Bs[tid * 8]) = o;
}

// ---- K8: V = X @ Wv^T + bv, output transposed bf16 Vt[h][n] ----------------
// block tile 128(M=n) x 64(N=h), K-step 32; wave w -> rows 16w..16w+15.
__global__ void k_gemm_v(const float* __restrict__ X, const float* __restrict__ Wv,
                         const float* __restrict__ bv, unsigned short* __restrict__ Vt) {
    __shared__ __align__(16) unsigned short As[128 * 32];  // [m][k]
    __shared__ __align__(16) unsigned short Bs[64 * 32];   // [h][k]
    int tid = threadIdx.x, wave = tid >> 5, lane = tid & 31;
    int m0 = blockIdx.x * 128, n0 = blockIdx.y * 64;
    v8f acc[4] = {};
    int khalfA = (lane < 16) ? 0 : 16;
    int khalfB = (lane < 16) ? 0 : 16;
    int Mloc = wave * 16 + (lane & 15);
    for (int k0 = 0; k0 < DD; k0 += 32) {
        __syncthreads();
        stage_A_f32(X,  DD, m0, k0, As, tid);
        stage_B_f32(Wv, DD, n0, k0, Bs, tid);
        if (k0 + 32 < DD)
            __builtin_prefetch(&X[(size_t)(m0 + (tid >> 1)) * DD + k0 + 32], 0, 3);
        __syncthreads();
        FragB a;
        #pragma unroll
        for (int i = 0; i < 8; ++i) {
            int koff = 2 * i + khalfA;
            a.u[i] = *reinterpret_cast<const unsigned*>(&As[Mloc * 32 + koff]);
        }
        #pragma unroll
        for (int t = 0; t < 4; ++t) {
            FragB b;
            int nloc = t * 16 + (lane & 15);
            #pragma unroll
            for (int j = 0; j < 8; ++j)
                b.u[j] = *reinterpret_cast<const unsigned*>(&Bs[nloc * 32 + khalfB + 2 * j]);
            acc[t] = __builtin_amdgcn_wmma_f32_16x16x32_bf16(
                false, a.v, false, b.v, (short)0, acc[t], false, false);
        }
    }
    #pragma unroll
    for (int t = 0; t < 4; ++t) {
        int Ng = n0 + t * 16 + (lane & 15);
        float bias = bv[Ng];
        #pragma unroll
        for (int r = 0; r < 8; ++r) {
            int Mg = m0 + wave * 16 + r + ((lane < 16) ? 0 : 8);
            Vt[(size_t)Ng * NN + Mg] = f32_to_bf16(acc[t][r] + bias);
        }
    }
}

// ---- K9: aggregation = AW @ V  (A = aw f32; B = Vt bf16 via async DMA) -----
__global__ void k_gemm_agg(const float* __restrict__ aw,
                           const unsigned short* __restrict__ Vt,
                           float* __restrict__ agg) {
    __shared__ __align__(16) unsigned short As[128 * 32];  // [g][k=n]
    __shared__ __align__(16) unsigned short Bs[64 * 32];   // [h][k=n]
    int tid = threadIdx.x, wave = tid >> 5, lane = tid & 31;
    int m0 = blockIdx.x * 128, n0 = blockIdx.y * 64;
    v8f acc[4] = {};
    int khalfA = (lane < 16) ? 0 : 16;
    int khalfB = (lane < 16) ? 0 : 16;
    int Mloc = wave * 16 + (lane & 15);
    // per-thread async copy slot: 16 bytes of the bf16 B tile (already bf16 in ws)
    int bhl = tid >> 2, bkl = (tid & 3) * 8;
    unsigned bs_off = (unsigned)(uintptr_t)(&Bs[tid * 8]);   // LDS byte offset (low 32 bits)
    for (int k0 = 0; k0 < NN; k0 += 32) {
        __syncthreads();
        stage_A_f32(aw, NN, m0, k0, As, tid);
        {   // CDNA5 async global->LDS DMA for B tile (no VGPR round-trip)
            unsigned long long gaddr =
                (unsigned long long)(const void*)(Vt + (size_t)(n0 + bhl) * NN + k0 + bkl);
            asm volatile("global_load_async_to_lds_b128 %0, %1, off"
                         :: "v"(bs_off), "v"(gaddr) : "memory");
            asm volatile("s_wait_asynccnt 0x0" ::: "memory");
        }
        if (k0 + 32 < NN)
            __builtin_prefetch(&aw[(size_t)(m0 + (tid >> 1)) * NN + k0 + 32], 0, 3);
        __syncthreads();
        FragB a;
        #pragma unroll
        for (int i = 0; i < 8; ++i) {
            int koff = 2 * i + khalfA;
            a.u[i] = *reinterpret_cast<const unsigned*>(&As[Mloc * 32 + koff]);
        }
        #pragma unroll
        for (int t = 0; t < 4; ++t) {
            FragB b;
            int nloc = t * 16 + (lane & 15);
            #pragma unroll
            for (int j = 0; j < 8; ++j)
                b.u[j] = *reinterpret_cast<const unsigned*>(&Bs[nloc * 32 + khalfB + 2 * j]);
            acc[t] = __builtin_amdgcn_wmma_f32_16x16x32_bf16(
                false, a.v, false, b.v, (short)0, acc[t], false, false);
        }
    }
    #pragma unroll
    for (int t = 0; t < 4; ++t) {
        int Ng = n0 + t * 16 + (lane & 15);
        #pragma unroll
        for (int r = 0; r < 8; ++r) {
            int Mg = m0 + wave * 16 + r + ((lane < 16) ? 0 : 8);
            agg[(size_t)Mg * HH + Ng] = acc[t][r];
        }
    }
}

extern "C" void kernel_launch(void* const* d_in, const int* in_sizes, int n_in,
                              void* d_out, int out_size, void* d_ws, size_t ws_size,
                              hipStream_t stream) {
    const float* X   = (const float*)d_in[0];
    const float* Wq  = (const float*)d_in[1];
    const float* bq  = (const float*)d_in[2];
    const float* Wk  = (const float*)d_in[3];
    const float* bk  = (const float*)d_in[4];
    const float* Wv  = (const float*)d_in[5];
    const float* bv  = (const float*)d_in[6];
    const float* sw  = (const float*)d_in[7];   // [N]
    const float* sb  = (const float*)d_in[8];   // scalar
    const int*   msk = (const int*)d_in[9];

    float* out = (float*)d_out;
    float* agg = out;                                   // [G,H]
    float* aw  = out + (size_t)GG * HH;                 // [G,N]

    float*    ws    = (float*)d_ws;
    float*    xtw   = ws;                  // 1024
    float*    kw    = ws + 1024;           // 1024
    float*    u     = ws + 2048;           // 1024
    float*    cscal = ws + 3072;           // 1
    float*    sumw  = ws + 3073;           // 1
    float*    s     = ws + 4096;           // 8192
    unsigned* mg    = (unsigned*)(ws + 12288);  // 2048
    float*    zg    = ws + 14336;          // 2048
    unsigned short* Vt = (unsigned short*)(ws + 16384); // [H][N] bf16, 16 MB

    k_init  <<<8,               256, 0, stream>>>(xtw, sumw, mg, zg);
    k_xtw   <<<64,              256, 0, stream>>>(X, sw, xtw, sumw);
    k_kw    <<<128,             256, 0, stream>>>(Wk, bk, xtw, sumw, kw);
    k_u     <<<4,               256, 0, stream>>>(Wq, bq, kw, sb, u, cscal);
    k_s     <<<1024,            256, 0, stream>>>(X, u, cscal, s);
    k_gemm_v<<<dim3(64, 16),    256, 0, stream>>>(X, Wv, bv, Vt);
    k_gmax  <<<dim3(8, 32),     256, 0, stream>>>(msk, s, mg);
    k_ewrite<<<dim3(32, 128),   256, 0, stream>>>(msk, s, mg, aw, zg);
    k_final <<<2048,            256, 0, stream>>>(aw, zg, mg);
    k_gemm_agg<<<dim3(16, 16),  256, 0, stream>>>(aw, Vt, agg);
}
